// Model_85366769975619
// MI455X (gfx1250) — matrix operand order repaired and checked by reading, not compiled, not run on hardware
//
#include <hip/hip_runtime.h>
#include <hip/hip_bf16.h>
#include <math.h>

// MI455X / gfx1250, wave32. One wave per sequence, all GEMMs via
// v_wmma_f32_16x16x32_f16 with f32 accumulate. Weights staged once per block
// into LDS in WMMA-B ("bt" = [N][K] with K zero-padded to 32) layout.

typedef _Float16 v16h __attribute__((ext_vector_type(16)));
typedef _Float16 v8h  __attribute__((ext_vector_type(8)));
typedef float    v8f  __attribute__((ext_vector_type(8)));

#define NT 17  // weight tiles: fw, 3x(qkv1,qkv2,wo,w1,w2), uw

__device__ __forceinline__ v8f wmma16(v16h a, v16h b, v8f c) {
  return __builtin_amdgcn_wmma_f32_16x16x32_f16(false, a, false, b, (short)0, c,
                                                false, false);
}

// A operand (16x32 f16, K<=16 used): lane L<16 holds row L, K=0..7 in halves
// 0..7; lane L>=16 holds row L-16, K=8..15 in halves 0..7; halves 8..15 (K>=16)
// are zero. Staging buffer: [16 rows][16 K] f16.
__device__ __forceinline__ v16h load_A(const _Float16* ast, int lane) {
  int row = lane & 15;
  int ks  = (lane < 16) ? 0 : 8;
  v8h lo = *(const v8h*)(ast + row * 16 + ks);
  v16h a;
#pragma unroll
  for (int i = 0; i < 8; ++i) { a[i] = lo[i]; a[i + 8] = (_Float16)0; }
  return a;
}

// B operand (32x16 f16): lane L<16 = column N=L, K=0..7 then 16..23; lane L>=16 =
// column N=L-16, K=8..15 then 24..31. Staging: bt[N=16][K=32] f16, K>=16 zeroed.
__device__ __forceinline__ v16h load_B(const _Float16* bt, int lane) {
  int n  = lane & 15;
  int kb = (lane < 16) ? 0 : 8;
  v8h lo = *(const v8h*)(bt + n * 32 + kb);
  v8h hi = *(const v8h*)(bt + n * 32 + 16 + kb);
  v16h r;
#pragma unroll
  for (int i = 0; i < 8; ++i) { r[i] = lo[i]; r[i + 8] = hi[i]; }
  return r;
}

// C/D layout: VGPR r, lanes 0-15 = (M=r, N=lane); lanes 16-31 = (M=r+8, N=lane-16).
__device__ __forceinline__ v8f bias_C(const float* bv, int lane) {
  float v = bv[lane & 15];
  v8f c;
#pragma unroll
  for (int r = 0; r < 8; ++r) c[r] = v;
  return c;
}

// C tile -> A staging: A[l][k] = C[l][k] for k < width, else 0.
__device__ __forceinline__ void stage_A_from_C(_Float16* ast, const v8f c,
                                               int lane, int width, float scale) {
  int col = lane & 15;
  int r0  = (lane < 16) ? 0 : 8;
  bool act = col < width;
#pragma unroll
  for (int r = 0; r < 8; ++r)
    ast[(r0 + r) * 16 + col] = (_Float16)(act ? c[r] * scale : 0.0f);
}

// Per-head masked A from qkv tile (q in cols 0..7): keep only k=2h,2h+1.
__device__ __forceinline__ void stage_A_head(_Float16* ast, const v8f c1,
                                             int lane, int h, float scale) {
  int col = lane & 15;
  int r0  = (lane < 16) ? 0 : 8;
  bool act = (col == 2 * h) || (col == 2 * h + 1);
#pragma unroll
  for (int r = 0; r < 8; ++r)
    ast[(r0 + r) * 16 + col] = (_Float16)(act ? c1[r] * scale : 0.0f);
}

// Build bt[j][k] = k_mat[j][k] (k=0..7) from qkv tile (k_mat in cols 8..15).
__device__ __forceinline__ void stage_BT_kT(_Float16* bst, const v8f c1, int lane) {
  int col = lane & 15;
  int r0  = (lane < 16) ? 0 : 8;
  if (col >= 8) {
#pragma unroll
    for (int r = 0; r < 8; ++r) bst[(r0 + r) * 32 + (col - 8)] = (_Float16)c1[r];
  } else {
#pragma unroll
    for (int r = 0; r < 8; ++r) bst[(r0 + r) * 32 + 8 + col] = (_Float16)0;
  }
}

// Build bt[n][j] = v[j][n] masked to n in {2h,2h+1} from v tile (cols 0..7).
__device__ __forceinline__ void stage_BT_vh(_Float16* bst, const v8f c2,
                                            int lane, int h) {
  int col = lane & 15;
  int r0  = (lane < 16) ? 0 : 8;
  bool act = (col == 2 * h) || (col == 2 * h + 1);
#pragma unroll
  for (int r = 0; r < 8; ++r)
    bst[col * 32 + (r0 + r)] = (_Float16)(act ? c2[r] : 0.0f);
}

__device__ __forceinline__ void softmax16(v8f& s) {
#pragma unroll
  for (int r = 0; r < 8; ++r) {
    float m = s[r];
#pragma unroll
    for (int d = 1; d < 16; d <<= 1) m = fmaxf(m, __shfl_xor(m, d, 32));
    float e = __expf(s[r] - m);
    float sum = e;
#pragma unroll
    for (int d = 1; d < 16; d <<= 1) sum += __shfl_xor(sum, d, 32);
    s[r] = e / sum;
  }
}

// x = LayerNorm(x + o) over cols 0..7; padded cols forced to 0.
__device__ __forceinline__ void layer_norm(v8f& x, const v8f o, const float* gv,
                                           const float* bv, int lane) {
  int col = lane & 15;
  float g = gv[col], bb = bv[col];
#pragma unroll
  for (int r = 0; r < 8; ++r) {
    float t  = x[r] + o[r];
    float s1 = t, s2 = t * t;
#pragma unroll
    for (int d = 1; d < 16; d <<= 1) {
      s1 += __shfl_xor(s1, d, 32);
      s2 += __shfl_xor(s2, d, 32);
    }
    float m  = s1 * 0.125f;
    float va = s2 * 0.125f - m * m;
    float rs = rsqrtf(va + 1e-5f);
    x[r] = (col < 8) ? (g * (t - m) * rs + bb) : 0.0f;
  }
}

__device__ __forceinline__ float wsrc(int tile, int n, int k, const float* fw,
                                      const float* ipw, const float* ow,
                                      const float* l1w, const float* l2w,
                                      const float* uw) {
  if (tile == 0)  return (n < 8 && k < 4) ? fw[n * 4 + k] : 0.f;
  if (tile == 16) return (n < 16 && k < 8) ? uw[n * 8 + k] : 0.f;
  int L = (tile - 1) / 5, kind = (tile - 1) % 5;
  switch (kind) {
    case 0:  return (n < 16 && k < 8) ? ipw[(L * 24 + n) * 8 + k] : 0.f;       // q|k
    case 1:  return (n < 8  && k < 8) ? ipw[(L * 24 + 16 + n) * 8 + k] : 0.f;  // v
    case 2:  return (n < 8  && k < 8) ? ow[(L * 8 + n) * 8 + k] : 0.f;
    case 3:  return (n < 16 && k < 8) ? l1w[(L * 16 + n) * 8 + k] : 0.f;
    default: return (n < 8  && k < 16) ? l2w[(L * 8 + n) * 16 + k] : 0.f;
  }
}

__device__ __forceinline__ float bsrc(int tile, int n, const float* fb,
                                      const float* ipb, const float* ob,
                                      const float* l1b, const float* l2b,
                                      const float* ub) {
  if (tile == 0)  return (n < 8) ? fb[n] : 0.f;
  if (tile == 16) return ub[n];
  int L = (tile - 1) / 5, kind = (tile - 1) % 5;
  switch (kind) {
    case 0:  return ipb[L * 24 + n];
    case 1:  return (n < 8) ? ipb[L * 24 + 16 + n] : 0.f;
    case 2:  return (n < 8) ? ob[L * 8 + n] : 0.f;
    case 3:  return l1b[L * 16 + n];
    default: return (n < 8) ? l2b[L * 8 + n] : 0.f;
  }
}

__global__ __launch_bounds__(256) void fused_tower_kernel(
    const int* cidxs, const int* slots, const float* gfeats, const float* f2,
    const float* f3, const float* tab_c, const float* tab_s, const float* fw,
    const float* fb, const float* ipw, const float* ipb, const float* ow,
    const float* ob, const float* l1w, const float* l1b, const float* l2w,
    const float* l2b, const float* n1g, const float* n1b, const float* n2g,
    const float* n2b, const float* uw, const float* ub, const float* w1,
    const float* b1, const float* w2, const float* b2, float* y, int* nf3,
    int K) {
  __shared__ __align__(64) _Float16 wbt[NT][512];   // weight B tiles [N=16][K=32]
  __shared__ __align__(64) _Float16 asc[8][256];    // per-wave A staging [16][16]
  __shared__ __align__(64) _Float16 bsc[8][512];    // per-wave B staging [16][32]
  __shared__ float biasv[NT][16];
  __shared__ float nrm[192];                        // [3][2 norms][g/b][16]
  __shared__ float floorbuf[8][8];
  __shared__ float ubuf[16];
  __shared__ float zb[40];
  __shared__ float hbuf[4][20];

  const int tid  = threadIdx.x;
  const int lane = tid & 31;
  const int wv   = tid >> 5;
  const int b    = blockIdx.x;
  const int seq  = b * 8 + wv;  // F = 8 floors

  // ---- Phase 0: stage weights/biases/norm params into LDS (whole block) ----
  for (int t = tid; t < NT * 512; t += 256) {
    int tile = t >> 9, idx = t & 511, n = idx >> 5, k = idx & 31;
    wbt[tile][idx] = (_Float16)wsrc(tile, n, k, fw, ipw, ow, l1w, l2w, uw);
  }
  for (int t = tid; t < NT * 16; t += 256)
    biasv[t >> 4][t & 15] = bsrc(t >> 4, t & 15, fb, ipb, ob, l1b, l2b, ub);
  for (int t = tid; t < 192; t += 256) {
    int L = t >> 6, rem = t & 63, which = rem >> 5, gb = (rem >> 4) & 1,
        col = rem & 15;
    const float* src = which ? (gb ? n2b : n2g) : (gb ? n1b : n1g);
    nrm[t] = (col < 8) ? src[L * 8 + col] : (gb ? 0.f : 1.f);
  }
  __syncthreads();

  _Float16* Ast = &asc[wv][0];
  _Float16* Bst = &bsc[wv][0];
  // zero the K=16..31 pad of the per-wave B staging once
  for (int t = lane; t < 256; t += 32) Bst[(t >> 4) * 32 + 16 + (t & 15)] = (_Float16)0;

  // ---- Phase 1: emb_floor  x = g @ fw^T + fb  (K=4) ----
  {
    int row = lane & 15;
    const float* g = gfeats + (size_t)seq * 64 + row * 4;
    if (lane < 16) {
#pragma unroll
      for (int c = 0; c < 4; ++c) Ast[row * 16 + c] = (_Float16)g[c];
#pragma unroll
      for (int c = 4; c < 8; ++c) Ast[row * 16 + c] = (_Float16)0;
    } else {
#pragma unroll
      for (int c = 8; c < 16; ++c) Ast[row * 16 + c] = (_Float16)0;
    }
  }
  v8f x = bias_C(biasv[0], lane);
  x = wmma16(load_A(Ast, lane), load_B(&wbt[0][0], lane), x);

  // ---- 3 transformer encoder layers ----
#pragma unroll
  for (int L = 0; L < 3; ++L) {
    const int t0 = 1 + L * 5;
    // qkv = x @ ipw^T + ipb   (two N-tiles: [q|k] and [v])
    stage_A_from_C(Ast, x, lane, 8, 1.0f);
    v16h Ax = load_A(Ast, lane);
    v8f c1 = bias_C(biasv[t0], lane);
    c1 = wmma16(Ax, load_B(&wbt[t0][0], lane), c1);
    v8f c2 = bias_C(biasv[t0 + 1], lane);
    c2 = wmma16(Ax, load_B(&wbt[t0 + 1][0], lane), c2);

    // attention: per head h, s = (q_h/sqrt(2)) @ k_h^T, softmax, o += a_h @ v_h
    stage_BT_kT(Bst, c1, lane);
    v16h Bk = load_B(Bst, lane);
    v8f ao;
#pragma unroll
    for (int r = 0; r < 8; ++r) ao[r] = 0.f;
#pragma unroll
    for (int h = 0; h < 4; ++h) {
      stage_A_head(Ast, c1, lane, h, 0.70710678f);
      v16h Aq = load_A(Ast, lane);
      v8f s;
#pragma unroll
      for (int r = 0; r < 8; ++r) s[r] = 0.f;
      s = wmma16(Aq, Bk, s);
      softmax16(s);
      stage_A_from_C(Ast, s, lane, 16, 1.0f);
      v16h Aa = load_A(Ast, lane);
      stage_BT_vh(Bst, c2, lane, h);
      ao = wmma16(Aa, load_B(Bst, lane), ao);
    }
    // out-proj + LN1
    stage_A_from_C(Ast, ao, lane, 8, 1.0f);
    v8f o = bias_C(biasv[t0 + 2], lane);
    o = wmma16(load_A(Ast, lane), load_B(&wbt[t0 + 2][0], lane), o);
    layer_norm(x, o, &nrm[(L * 4 + 0) * 16], &nrm[(L * 4 + 1) * 16], lane);

    // FFN 8 -> 16 (relu) -> 8, then LN2
    stage_A_from_C(Ast, x, lane, 8, 1.0f);
    v8f h1 = bias_C(biasv[t0 + 3], lane);
    h1 = wmma16(load_A(Ast, lane), load_B(&wbt[t0 + 3][0], lane), h1);
#pragma unroll
    for (int r = 0; r < 8; ++r) h1[r] = fmaxf(h1[r], 0.f);
    stage_A_from_C(Ast, h1, lane, 16, 1.0f);
    v8f h2 = bias_C(biasv[t0 + 4], lane);
    h2 = wmma16(load_A(Ast, lane), load_B(&wbt[t0 + 4][0], lane), h2);
    layer_norm(x, h2, &nrm[(L * 4 + 2) * 16], &nrm[(L * 4 + 3) * 16], lane);
  }

  // ---- floors: sum over L (rows) ----
  {
    float v = 0.f;
#pragma unroll
    for (int r = 0; r < 8; ++r) v += x[r];
    v += __shfl_xor(v, 16, 32);
    if (lane < 8) floorbuf[wv][lane] = v;
  }
  __syncthreads();

  // ---- units = sum_f relu(floors_f @ uw^T + ub) : one WMMA on wave 0 ----
  if (wv == 0) {
    int col = lane & 15, r0 = (lane < 16) ? 0 : 8;
#pragma unroll
    for (int r = 0; r < 8; ++r) {
      int row = r0 + r;
      asc[0][row * 16 + col] =
          (row < 8 && col < 8) ? (_Float16)floorbuf[row][col] : (_Float16)0;
    }
    v8f u = bias_C(biasv[16], lane);
    u = wmma16(load_A(&asc[0][0], lane), load_B(&wbt[16][0], lane), u);
    float t = 0.f;
#pragma unroll
    for (int r = 0; r < 8; ++r) t += fmaxf(u[r], 0.f);  // rows f=0..7 (low half)
    if (lane < 16) ubuf[lane] = t;
  }
  __syncthreads();

  // ---- f1f2 = [tab_c | tab_s | units | f2] (36) ----
  if (tid < 8)                zb[tid] = tab_c[(size_t)cidxs[b] * 8 + tid];
  else if (tid < 16)          zb[tid] = tab_s[(size_t)slots[b] * 8 + (tid - 8)];
  else if (tid < 32)          zb[tid] = ubuf[tid - 16];
  else if (tid < 36)          zb[tid] = f2[b * 4 + (tid - 32)];
  __syncthreads();

  // ---- ff_match: h = relu(z @ w1^T + b1); y = h @ w2^T + b2 ----
  if (tid < K * 20) {
    int k = tid / 20, j = tid - 20 * k;
    float acc = b1[j];
    const float* wr = w1 + j * 42;
#pragma unroll
    for (int c = 0; c < 36; ++c) acc += zb[c] * wr[c];
    const float* f3p = f3 + (size_t)(b * K + k) * 6;
#pragma unroll
    for (int c = 0; c < 6; ++c) acc += f3p[c] * wr[36 + c];
    hbuf[k][j] = fmaxf(acc, 0.f);
  }
  __syncthreads();
  if (tid < K) {
    float acc = b2[0];
#pragma unroll
    for (int j = 0; j < 20; ++j) acc += hbuf[tid][j] * w2[j];
    y[(size_t)b * K + tid] = acc;
  }
  if (tid == 0) nf3[b] = K;
}

extern "C" void kernel_launch(void* const* d_in, const int* in_sizes, int n_in,
                              void* d_out, int out_size, void* d_ws, size_t ws_size,
                              hipStream_t stream) {
  const int*   cidxs = (const int*)d_in[0];
  const int*   slots = (const int*)d_in[1];
  const float* gfeats = (const float*)d_in[2];
  const float* f2  = (const float*)d_in[3];
  const float* f3  = (const float*)d_in[4];
  const float* tab_c = (const float*)d_in[5];
  const float* tab_s = (const float*)d_in[6];
  const float* fw  = (const float*)d_in[7];
  const float* fb  = (const float*)d_in[8];
  const float* ipw = (const float*)d_in[9];
  const float* ipb = (const float*)d_in[10];
  const float* ow  = (const float*)d_in[11];
  const float* ob  = (const float*)d_in[12];
  const float* l1w = (const float*)d_in[13];
  const float* l1b = (const float*)d_in[14];
  const float* l2w = (const float*)d_in[15];
  const float* l2b = (const float*)d_in[16];
  const float* n1g = (const float*)d_in[17];
  const float* n1b = (const float*)d_in[18];
  const float* n2g = (const float*)d_in[19];
  const float* n2b = (const float*)d_in[20];
  const float* uw  = (const float*)d_in[21];
  const float* ub  = (const float*)d_in[22];
  const float* w1  = (const float*)d_in[23];
  const float* b1  = (const float*)d_in[24];
  const float* w2  = (const float*)d_in[25];
  const float* b2  = (const float*)d_in[26];

  const int B = in_sizes[0];
  const int K = in_sizes[4] / (B * 6);
  float* y = (float*)d_out;
  int* nf3 = (int*)(y + (size_t)B * K);

  fused_tower_kernel<<<B, 256, 0, stream>>>(
      cidxs, slots, gfeats, f2, f3, tab_c, tab_s, fw, fb, ipw, ipb, ow, ob, l1w,
      l1b, l2w, l2b, n1g, n1b, n2g, n2b, uw, ub, w1, b1, w2, b2, y, nf3, K);
}
